// Experts_9139690406408
// MI455X (gfx1250) — matrix operand order repaired and checked by reading, not compiled, not run on hardware
//
#include <hip/hip_runtime.h>

// Problem constants (match reference)
#define NE      64
#define DMODEL  1024
#define DFF     2048
#define CTOK    512   // tokens per expert

// Tile config: 128x128 block tile, K-step 32, 256 threads = 8 waves (wave32).
// Wave grid 4x2 -> each wave computes 32 rows x 64 cols = 2x4 WMMA 16x16 tiles.
// LDS double-buffered; rows padded to avoid bank conflicts on fragment reads.
#define PADA 20   // dwords per A row  (16 data + 4 pad)  -> lane stride 80B
#define PADB 68   // dwords per B row  (64 data + 4 pad)  -> lane stride 272B

typedef __attribute__((ext_vector_type(16))) __bf16 v16bf;
typedef __attribute__((ext_vector_type(2)))  __bf16 v2bf;
typedef __attribute__((ext_vector_type(8)))  float  v8f;
typedef __attribute__((ext_vector_type(4)))  unsigned int v4u;

union FragU {
    v16bf bf;
    v4u   q[2];
};

__device__ __forceinline__ unsigned pack2(float lo, float hi) {
    // native f32->bf16 (RNE); backend can fuse into v_cvt_pk_bf16_f32
    union { v2bf v; unsigned u; } c;
    c.v = (v2bf){(__bf16)lo, (__bf16)hi};
    return c.u;
}

__device__ __forceinline__ unsigned short bf16_bits(float f) {
    union { __bf16 b; unsigned short s; } c;
    c.b = (__bf16)f;
    return c.s;
}

__device__ __forceinline__ float gelu_tanh(float x) {
    // jax.nn.gelu default (approximate=True)
    float x3 = x * x * x;
    float t  = tanhf(0.7978845608028654f * (x + 0.044715f * x3));
    return 0.5f * x * (1.0f + t);
}

// ---------------------------------------------------------------------------
// GEMM1 + GELU:  h[e,c,f] = gelu( x[e,c,:] @ w1[e,:,f] ), h stored bf16 in ws
// grid: (DFF/128, CTOK/128, NE)
// ---------------------------------------------------------------------------
__global__ __launch_bounds__(256)
void moe_gemm1_gelu(const float* __restrict__ x,
                    const float* __restrict__ w1,
                    unsigned short* __restrict__ h) {
    __shared__ unsigned ldsA[2][128 * PADA];
    __shared__ unsigned ldsB[2][32 * PADB];

    const int tid   = threadIdx.x;
    const int lane  = tid & 31;
    const int wave  = tid >> 5;
    const int wm    = wave >> 1;        // 0..3 : 32-row band
    const int wn    = wave & 1;         // 0..1 : 64-col band

    const int e  = blockIdx.z;
    const int tm = blockIdx.y;
    const int tn = blockIdx.x;

    const float* Abase = x  + (size_t)(e * CTOK + tm * 128) * DMODEL;
    const float* Bbase = w1 + (size_t)e * DMODEL * DFF + (size_t)tn * 128;

    const int arow = tid >> 1;          // 0..127
    const int acol = (tid & 1) * 16;    // 0 / 16 floats
    const int brow = tid >> 3;          // 0..31
    const int bcol = (tid & 7) * 16;    // 0..112 floats

    const float* aptr = Abase + (size_t)arow * DMODEL + acol;
    const float* bptr = Bbase + (size_t)brow * DFF + bcol;

    const int laneM = lane & 15;
    const int kb    = (lane >> 4) << 3; // 0 or 8 (K offset for upper lane half)

    v8f acc[2][4];
#pragma unroll
    for (int t = 0; t < 2; ++t)
#pragma unroll
        for (int u = 0; u < 4; ++u)
            acc[t][u] = (v8f)(0.0f);

    const int NK = DMODEL / 32;

    // ---- prologue: tile 0 -> LDS buf 0 ----
    {
        const float4* pa = (const float4*)aptr;
        float4 a0 = pa[0], a1 = pa[1], a2 = pa[2], a3 = pa[3];
        const float4* pb = (const float4*)bptr;
        float4 b0 = pb[0], b1 = pb[1], b2 = pb[2], b3 = pb[3];
        unsigned* da = &ldsA[0][arow * PADA + (acol >> 1)];
        da[0] = pack2(a0.x, a0.y); da[1] = pack2(a0.z, a0.w);
        da[2] = pack2(a1.x, a1.y); da[3] = pack2(a1.z, a1.w);
        da[4] = pack2(a2.x, a2.y); da[5] = pack2(a2.z, a2.w);
        da[6] = pack2(a3.x, a3.y); da[7] = pack2(a3.z, a3.w);
        unsigned* db = &ldsB[0][brow * PADB + (bcol >> 1)];
        db[0] = pack2(b0.x, b0.y); db[1] = pack2(b0.z, b0.w);
        db[2] = pack2(b1.x, b1.y); db[3] = pack2(b1.z, b1.w);
        db[4] = pack2(b2.x, b2.y); db[5] = pack2(b2.z, b2.w);
        db[6] = pack2(b3.x, b3.y); db[7] = pack2(b3.z, b3.w);
    }
    __syncthreads();

    int buf = 0;
    for (int kt = 0; kt < NK; ++kt) {
        const bool next = (kt + 1) < NK;

        // ---- issue next tile's global loads early (hide HBM latency) ----
        float4 na0{}, na1{}, na2{}, na3{}, nb0{}, nb1{}, nb2{}, nb3{};
        if (next) {
            const float4* pa = (const float4*)(aptr + (kt + 1) * 32);
            na0 = pa[0]; na1 = pa[1]; na2 = pa[2]; na3 = pa[3];
            const float4* pb = (const float4*)(bptr + (size_t)(kt + 1) * 32 * DFF);
            nb0 = pb[0]; nb1 = pb[1]; nb2 = pb[2]; nb3 = pb[3];
            if (kt + 2 < NK) {
                __builtin_prefetch(aptr + (kt + 2) * 32, 0, 1);
                __builtin_prefetch(bptr + (size_t)(kt + 2) * 32 * DFF, 0, 1);
            }
        }

        // ---- LDS -> fragments -> WMMA ----
        FragU a[2];
#pragma unroll
        for (int t = 0; t < 2; ++t) {
            const unsigned* src = &ldsA[buf][(wm * 32 + t * 16 + laneM) * PADA];
            a[t].q[0] = *(const v4u*)(src + (kb >> 1));          // K = kb .. kb+7
            a[t].q[1] = *(const v4u*)(src + ((kb + 16) >> 1));   // K = kb+16 .. kb+23
        }
        FragU b[4];
#pragma unroll
        for (int u = 0; u < 4; ++u) {
            const unsigned short* src = (const unsigned short*)&ldsB[buf][(wn * 64 + u * 16) >> 1] + laneM;
#pragma unroll
            for (int s = 0; s < 16; ++s)
                b[u].bf[s] = __builtin_bit_cast(__bf16, src[(kb + (s & 7) + ((s >> 3) << 4)) * (2 * PADB)]);
        }
#pragma unroll
        for (int t = 0; t < 2; ++t)
#pragma unroll
            for (int u = 0; u < 4; ++u)
                acc[t][u] = __builtin_amdgcn_wmma_f32_16x16x32_bf16(
                    false, a[t].bf, false, b[u].bf, (short)0, acc[t][u], false, false);

        // ---- pack+store next tile into the other buffer ----
        if (next) {
            unsigned* da = &ldsA[buf ^ 1][arow * PADA + (acol >> 1)];
            da[0] = pack2(na0.x, na0.y); da[1] = pack2(na0.z, na0.w);
            da[2] = pack2(na1.x, na1.y); da[3] = pack2(na1.z, na1.w);
            da[4] = pack2(na2.x, na2.y); da[5] = pack2(na2.z, na2.w);
            da[6] = pack2(na3.x, na3.y); da[7] = pack2(na3.z, na3.w);
            unsigned* db = &ldsB[buf ^ 1][brow * PADB + (bcol >> 1)];
            db[0] = pack2(nb0.x, nb0.y); db[1] = pack2(nb0.z, nb0.w);
            db[2] = pack2(nb1.x, nb1.y); db[3] = pack2(nb1.z, nb1.w);
            db[4] = pack2(nb2.x, nb2.y); db[5] = pack2(nb2.z, nb2.w);
            db[6] = pack2(nb3.x, nb3.y); db[7] = pack2(nb3.z, nb3.w);
            __syncthreads();
            buf ^= 1;
        }
    }

    // ---- epilogue: GELU -> bf16 h ----
    const int rbase = (lane >> 4) * 8;   // C/D layout: upper lane half = M+8
#pragma unroll
    for (int t = 0; t < 2; ++t) {
        const int rowl = tm * 128 + wm * 32 + t * 16 + rbase;
#pragma unroll
        for (int u = 0; u < 4; ++u) {
            const int col = tn * 128 + wn * 64 + u * 16 + laneM;
            unsigned short* dst = h + (size_t)(e * CTOK + rowl) * DFF + col;
#pragma unroll
            for (int r = 0; r < 8; ++r)
                dst[(size_t)r * DFF] = bf16_bits(gelu_tanh(acc[t][u][r]));
        }
    }
}

// ---------------------------------------------------------------------------
// GEMM2:  y[e,c,d] = h[e,c,:] @ w2[e,:,d]   (h bf16 in ws, y fp32 out)
// A operand is already bf16 -> copy global->LDS with CDNA5 async-to-LDS ops.
// grid: (DMODEL/128, CTOK/128, NE)
// ---------------------------------------------------------------------------
__global__ __launch_bounds__(256)
void moe_gemm2(const unsigned short* __restrict__ h,
               const float* __restrict__ w2,
               float* __restrict__ y) {
    __shared__ unsigned ldsA[2][128 * PADA];
    __shared__ unsigned ldsB[2][32 * PADB];

    const int tid   = threadIdx.x;
    const int lane  = tid & 31;
    const int wave  = tid >> 5;
    const int wm    = wave >> 1;
    const int wn    = wave & 1;

    const int e  = blockIdx.z;
    const int tm = blockIdx.y;
    const int tn = blockIdx.x;

    const unsigned short* Abase = h + (size_t)(e * CTOK + tm * 128) * DFF;
    const float*          Bbase = w2 + (size_t)e * DFF * DMODEL + (size_t)tn * 128;

    const int arow = tid >> 1;
    const int acol = (tid & 1) * 16;     // bf16 units
    const int brow = tid >> 3;
    const int bcol = (tid & 7) * 16;     // floats

    const unsigned short* agptr = Abase + (size_t)arow * DFF + acol;
    const float*          bptr  = Bbase + (size_t)brow * DMODEL + bcol;

    // LDS byte offsets for the async destination (flat addr low 32 bits = LDS offset)
    unsigned ldsa_off[2];
    ldsa_off[0] = (unsigned)(size_t)&ldsA[0][arow * PADA + (acol >> 1)];
    ldsa_off[1] = (unsigned)(size_t)&ldsA[1][arow * PADA + (acol >> 1)];

    const int laneM = lane & 15;
    const int kb    = (lane >> 4) << 3;

    v8f acc[2][4];
#pragma unroll
    for (int t = 0; t < 2; ++t)
#pragma unroll
        for (int u = 0; u < 4; ++u)
            acc[t][u] = (v8f)(0.0f);

    const int NK = DFF / 32;

    // ---- prologue: tile 0 ----
    {
        unsigned long long ga = (unsigned long long)(size_t)agptr;
        asm volatile("global_load_async_to_lds_b128 %0, %1, off"
                     :: "v"(ldsa_off[0]), "v"(ga) : "memory");
        asm volatile("global_load_async_to_lds_b128 %0, %1, off offset:16"
                     :: "v"(ldsa_off[0]), "v"(ga) : "memory");

        const float4* pb = (const float4*)bptr;
        float4 b0 = pb[0], b1 = pb[1], b2 = pb[2], b3 = pb[3];
        unsigned* db = &ldsB[0][brow * PADB + (bcol >> 1)];
        db[0] = pack2(b0.x, b0.y); db[1] = pack2(b0.z, b0.w);
        db[2] = pack2(b1.x, b1.y); db[3] = pack2(b1.z, b1.w);
        db[4] = pack2(b2.x, b2.y); db[5] = pack2(b2.z, b2.w);
        db[6] = pack2(b3.x, b3.y); db[7] = pack2(b3.z, b3.w);
        asm volatile("s_wait_asynccnt 0x0" ::: "memory");
    }
    __syncthreads();

    int buf = 0;
    for (int kt = 0; kt < NK; ++kt) {
        const bool next = (kt + 1) < NK;

        float4 nb0{}, nb1{}, nb2{}, nb3{};
        if (next) {
            // async A tile (bf16, no conversion) straight into the other buffer
            unsigned long long ga =
                (unsigned long long)(size_t)(agptr + (kt + 1) * 32);
            asm volatile("global_load_async_to_lds_b128 %0, %1, off"
                         :: "v"(ldsa_off[buf ^ 1]), "v"(ga) : "memory");
            asm volatile("global_load_async_to_lds_b128 %0, %1, off offset:16"
                         :: "v"(ldsa_off[buf ^ 1]), "v"(ga) : "memory");

            const float4* pb = (const float4*)(bptr + (size_t)(kt + 1) * 32 * DMODEL);
            nb0 = pb[0]; nb1 = pb[1]; nb2 = pb[2]; nb3 = pb[3];
            if (kt + 2 < NK)
                __builtin_prefetch(bptr + (size_t)(kt + 2) * 32 * DMODEL, 0, 1);
        }

        FragU a[2];
#pragma unroll
        for (int t = 0; t < 2; ++t) {
            const unsigned* src = &ldsA[buf][(wm * 32 + t * 16 + laneM) * PADA];
            a[t].q[0] = *(const v4u*)(src + (kb >> 1));
            a[t].q[1] = *(const v4u*)(src + ((kb + 16) >> 1));
        }
        FragU b[4];
#pragma unroll
        for (int u = 0; u < 4; ++u) {
            const unsigned short* src = (const unsigned short*)&ldsB[buf][(wn * 64 + u * 16) >> 1] + laneM;
#pragma unroll
            for (int s = 0; s < 16; ++s)
                b[u].bf[s] = __builtin_bit_cast(__bf16, src[(kb + (s & 7) + ((s >> 3) << 4)) * (2 * PADB)]);
        }
#pragma unroll
        for (int t = 0; t < 2; ++t)
#pragma unroll
            for (int u = 0; u < 4; ++u)
                acc[t][u] = __builtin_amdgcn_wmma_f32_16x16x32_bf16(
                    false, a[t].bf, false, b[u].bf, (short)0, acc[t][u], false, false);

        if (next) {
            unsigned* db = &ldsB[buf ^ 1][brow * PADB + (bcol >> 1)];
            db[0] = pack2(nb0.x, nb0.y); db[1] = pack2(nb0.z, nb0.w);
            db[2] = pack2(nb1.x, nb1.y); db[3] = pack2(nb1.z, nb1.w);
            db[4] = pack2(nb2.x, nb2.y); db[5] = pack2(nb2.z, nb2.w);
            db[6] = pack2(nb3.x, nb3.y); db[7] = pack2(nb3.z, nb3.w);
            asm volatile("s_wait_asynccnt 0x0" ::: "memory");
            __syncthreads();
            buf ^= 1;
        }
    }

    const int rbase = (lane >> 4) * 8;
#pragma unroll
    for (int t = 0; t < 2; ++t) {
        const int rowl = tm * 128 + wm * 32 + t * 16 + rbase;
#pragma unroll
        for (int u = 0; u < 4; ++u) {
            const int col = tn * 128 + wn * 64 + u * 16 + laneM;
            float* dst = y + (size_t)(e * CTOK + rowl) * DMODEL + col;
#pragma unroll
            for (int r = 0; r < 8; ++r)
                dst[(size_t)r * DMODEL] = acc[t][u][r];
        }
    }
}

// ---------------------------------------------------------------------------
extern "C" void kernel_launch(void* const* d_in, const int* in_sizes, int n_in,
                              void* d_out, int out_size, void* d_ws, size_t ws_size,
                              hipStream_t stream) {
    const float* x  = (const float*)d_in[0];   // [32768, 1024]
    const float* w1 = (const float*)d_in[1];   // [64, 1024, 2048]
    const float* w2 = (const float*)d_in[2];   // [64, 2048, 1024]
    float* y = (float*)d_out;                  // [32768, 1024]

    // bf16 intermediate h: 32768 * 2048 * 2 bytes = 128 MiB in workspace
    unsigned short* h = (unsigned short*)d_ws;

    dim3 blk(256, 1, 1);
    dim3 g1(DFF / 128, CTOK / 128, NE);      // (16, 4, 64)
    dim3 g2(DMODEL / 128, CTOK / 128, NE);   // (8, 4, 64)

    moe_gemm1_gelu<<<g1, blk, 0, stream>>>(x, w1, h);
    moe_gemm2<<<g2, blk, 0, stream>>>(h, w2, y);
}
